// CvtLstm_60593398612147
// MI455X (gfx1250) — matrix operand order repaired and checked by reading, not compiled, not run on hardware
//
#include <hip/hip_runtime.h>

typedef __attribute__((ext_vector_type(16))) _Float16 v16h;
typedef __attribute__((ext_vector_type(8)))  float    v8f;

#define DEV static __device__ __forceinline__

// ---------------------------------------------------------------------------
// WMMA fragment helpers (CDNA5 wave32, 16x16x32 f16 -> f32)
//
// A (16x32, MxK) lane layout: lanes 0-15 row M=l, K {0..7} in v[0..7],
//   K {16..23} in v[8..15]; lanes 16-31 row M=l-16, K {8..15}, {24..31}.
// B (32x16, KxN) lane layout: lanes 0-15 col N=l, K {0..7} in v[0..7],
//   K {16..23} in v[8..15]; lanes 16-31 col N=l-16, K {8..15}, {24..31}.
// C/D (16x16 f32): lane l col N=l%16, v[j] row M = j + 8*(l/16).
// ---------------------------------------------------------------------------
DEV v16h wmma_ld2(const _Float16* p0, const _Float16* p1) {
  union U { v16h v; float4 f[2]; } u;
  u.f[0] = *(const float4*)p0;
  u.f[1] = *(const float4*)p1;
  return u.v;
}

// A stored [m][k], k contiguous. base = &A[mbase*lda + k0]
DEV v16h load_a(const _Float16* base, int lda) {
  const int l = threadIdx.x & 31;
  const _Float16* p = base + (l & 15) * lda + ((l >> 4) << 3);
  return wmma_ld2(p, p + 16);
}

// B source stored [n][k], k contiguous. base = &B[nbase*ldb + k0]
DEV v16h load_b(const _Float16* base, int ldb) {
  const int l = threadIdx.x & 31;
  const _Float16* p = base + (l & 15) * ldb + ((l >> 4) << 3);
  return wmma_ld2(p, p + 16);
}

DEV v8f wmma32(v16h a, v16h b, v8f c) {
  return __builtin_amdgcn_wmma_f32_16x16x32_f16(false, a, false, b,
                                                (short)0, c, false, false);
}

// ---------------------------------------------------------------------------
// Problem constants
// ---------------------------------------------------------------------------
#define NB    32
#define HW    256
#define MTOT  8192      // NB*HW
#define RCH   128
#define CM    128
#define ACH   128
#define ICH   64

// ---------------------------------------------------------------------------
// Conversion / staging kernels (fp32 -> f16, layout reshuffles)
// ---------------------------------------------------------------------------
__global__ void k_cvt_nchw(const float* __restrict__ src, _Float16* __restrict__ dst,
                           int C, int total) {
  for (int i = blockIdx.x * blockDim.x + threadIdx.x; i < total;
       i += gridDim.x * blockDim.x) {
    int m = i / C, c = i - m * C;
    int nb = m >> 8, hw = m & 255;
    dst[i] = (_Float16)src[((size_t)nb * C + c) * HW + hw];
  }
}

__global__ void k_cvt_flat(const float* __restrict__ s, _Float16* __restrict__ d, int n) {
  for (int i = blockIdx.x * blockDim.x + threadIdx.x; i < n;
       i += gridDim.x * blockDim.x)
    d[i] = (_Float16)s[i];
}

// W[o][c][3][3] -> d[o][(ky*3+kx)*128 + c]
__global__ void k_cvt_conv(const float* __restrict__ W, _Float16* __restrict__ d, int total) {
  for (int i = blockIdx.x * blockDim.x + threadIdx.x; i < total;
       i += gridDim.x * blockDim.x) {
    int o = i / 1152, r = i - o * 1152;
    int seg = r >> 7, c = r & 127;
    int ky = seg / 3, kx = seg - ky * 3;
    d[i] = (_Float16)W[(((size_t)o * 128 + c) * 3 + ky) * 3 + kx];
  }
}

// WG[gi*128+o][ b*128+c | 512+c (skip xc) | 640+c (skip hc) ]  (512 x 768)
__global__ void k_cvt_tok(const float* __restrict__ Wtok, const float* __restrict__ Wskip,
                          _Float16* __restrict__ d) {
  const int total = 512 * 768;
  for (int i = blockIdx.x * blockDim.x + threadIdx.x; i < total;
       i += gridDim.x * blockDim.x) {
    int og = i / 768, k = i - og * 768;
    int gi = og >> 7, o = og & 127;
    float v;
    if (k < 512) {
      int b = k >> 7, c = k & 127;
      v = Wtok[(((size_t)gi * 4 + b) * 128 + o) * 128 + c];
    } else if (k < 640) {
      v = Wskip[(((size_t)gi * 2 + 0) * 128 + o) * 128 + (k - 512)];
    } else {
      v = Wskip[(((size_t)gi * 2 + 1) * 128 + o) * 128 + (k - 640)];
    }
    d[i] = (_Float16)v;
  }
}

// ---------------------------------------------------------------------------
// Generic WMMA GEMM: C[M,N] = A[M,K] * B^T, A [m][k], B [n][k], both f16.
// Block = 8 waves, tile 32(M) x 64(N); wave tile 16x16.
// mode: 0 in-proj (bias+tanh -> f16 [m][128])
//       2 q/k proj (f16 scatter to [(b,n,g),hw,cc]),  aux = branch
//       3 v proj   (f16 scatter to [(b,n,g),cc,hw]),  aux = branch
//       4 gates    (f32 [m][512])
//       5 out-proj (bias, f32 NCHW)
// ---------------------------------------------------------------------------
__global__ __launch_bounds__(256) void k_gemm(const _Float16* __restrict__ A, int lda,
                                              const _Float16* __restrict__ B, int ldb,
                                              int K, void* __restrict__ out,
                                              const float* __restrict__ bias,
                                              int mode, int aux) {
  const int wave = threadIdx.x >> 5;
  const int mbase = blockIdx.x * 32 + (wave >> 2) * 16;
  const int nbase = blockIdx.y * 64 + (wave & 3) * 16;
  const _Float16* Ab = A + (size_t)mbase * lda;
  const _Float16* Bb = B + (size_t)nbase * ldb;
  v8f acc = {};
  for (int k = 0; k < K; k += 32) {
    v16h af = load_a(Ab + k, lda);
    v16h bf = load_b(Bb + k, ldb);
    acc = wmma32(af, bf, acc);
  }
  const int l = threadIdx.x & 31;
  const int col = nbase + (l & 15);
  const int rb = mbase + ((l >> 4) << 3);
  const float bv = bias ? bias[col] : 0.f;
#pragma unroll
  for (int j = 0; j < 8; ++j) {
    const int row = rb + j;
    float v = acc[j] + bv;
    if (mode == 0) {
      ((_Float16*)out)[(size_t)row * 128 + col] = (_Float16)tanhf(v);
    } else if (mode == 2) {
      int nb = row >> 8, hw = row & 255;
      int g = col >> 5, cc = col & 31;
      ((_Float16*)out)[((((size_t)aux * 32 + nb) * 4 + g) * 256 + hw) * 32 + cc] =
          (_Float16)v;
    } else if (mode == 3) {
      int nb = row >> 8, hw = row & 255;
      int g = col >> 5, cc = col & 31;
      ((_Float16*)out)[((((size_t)aux * 32 + nb) * 4 + g) * 32 + cc) * 256 + hw] =
          (_Float16)v;
    } else if (mode == 4) {
      ((float*)out)[(size_t)row * 512 + col] = v;
    } else { // 5
      int nb = row >> 8, hw = row & 255;
      ((float*)out)[((size_t)nb * 128 + col) * 256 + hw] = v;
    }
  }
}

// ---------------------------------------------------------------------------
// 3x3 conv as implicit GEMM: SRC f16 [m][128], B f16 [o][1152] ((ky,kx,c)),
// K = 36 steps of 32. Output f16 into ACT[m][768] at column dst_coff.
// ---------------------------------------------------------------------------
__global__ __launch_bounds__(256) void k_gemm_conv(const _Float16* __restrict__ SRC,
                                                   const _Float16* __restrict__ B,
                                                   _Float16* __restrict__ ACT,
                                                   int dst_coff) {
  const int wave = threadIdx.x >> 5;
  const int mbase = blockIdx.x * 32 + (wave >> 2) * 16;
  const int nbase = blockIdx.y * 64 + (wave & 3) * 16;
  const int l = threadIdx.x & 31;
  const int row_m = mbase + (l & 15);
  const int nb = row_m >> 8, hw = row_m & 255;
  const int h = hw >> 4, w = hw & 15;
  const int hi = l >> 4;
  const _Float16* Bb = B + (size_t)nbase * 1152;
  v8f acc = {};
  for (int step = 0; step < 36; ++step) {
    const int seg = step >> 2;
    const int coff = (step & 3) * 32;
    const int ky = seg / 3, kx = seg - ky * 3;
    const int h2 = h + ky - 1, w2 = w + kx - 1;
    v16h af;
#pragma unroll
    for (int e = 0; e < 16; ++e) af[e] = (_Float16)0.f;
    if ((unsigned)h2 < 16u && (unsigned)w2 < 16u) {
      const _Float16* p =
          SRC + ((size_t)(nb * 256 + h2 * 16 + w2)) * 128 + coff + hi * 8;
      af = wmma_ld2(p, p + 16);
    }
    v16h bf = load_b(Bb + step * 32, 1152);
    acc = wmma32(af, bf, acc);
  }
  const int col = nbase + (l & 15);
  const int rb = mbase + (hi << 3);
#pragma unroll
  for (int j = 0; j < 8; ++j)
    ACT[(size_t)(rb + j) * 768 + dst_coff + col] = (_Float16)acc[j];
}

// ---------------------------------------------------------------------------
// Flash attention, one (branch,batch,head) per blockIdx.y, wave = 16 q rows.
// q/k stored [hw][cc] (cc contiguous), v stored [cc][hw] (hw contiguous).
// Output scattered (f16) into ACT columns 0..511 at k = b*128 + g*32 + cc.
// ---------------------------------------------------------------------------
__global__ __launch_bounds__(256) void k_attn(const _Float16* __restrict__ QB,
                                              const _Float16* __restrict__ KB,
                                              const _Float16* __restrict__ VB,
                                              _Float16* __restrict__ ACT) {
  const int bng = blockIdx.y;           // ((b*32+n)*4+g)
  const int b = bng >> 7;
  const int n = (bng >> 2) & 31;
  const int g = bng & 3;
  const _Float16* q = QB + (size_t)bng * 256 * 32;
  const _Float16* kmat = KB + (size_t)bng * 256 * 32;
  const _Float16* vmat = VB + (size_t)bng * 32 * 256;

  const int wave = threadIdx.x >> 5;
  const int l = threadIdx.x & 31;
  const int hi = l >> 4;
  const int qbase = blockIdx.x * 128 + wave * 16;

  __shared__ __align__(16) _Float16 plds[8][16][32];

  // Q fragment: rows qbase..qbase+15, full K=32 (HC)
  v16h afq = load_a(q + (size_t)qbase * 32, 32);

  v8f acc0 = {}, acc1 = {};
  float m_r[8], l_r[8];
#pragma unroll
  for (int j = 0; j < 8; ++j) { m_r[j] = -1e30f; l_r[j] = 0.f; }

  for (int d0 = 0; d0 < 256; d0 += 32) {
    v16h bk0 = load_b(kmat + (size_t)d0 * 32, 32);
    v16h bk1 = load_b(kmat + (size_t)(d0 + 16) * 32, 32);
    v8f z = {};
    v8f s0 = wmma32(afq, bk0, z);
    v8f s1 = wmma32(afq, bk1, z);

    float p0[8], p1[8];
#pragma unroll
    for (int j = 0; j < 8; ++j) {
      float t = fmaxf(s0[j], s1[j]);
      t = fmaxf(t, __shfl_xor(t, 1));
      t = fmaxf(t, __shfl_xor(t, 2));
      t = fmaxf(t, __shfl_xor(t, 4));
      t = fmaxf(t, __shfl_xor(t, 8));
      const float mn = fmaxf(m_r[j], t);
      const float sc = __expf(m_r[j] - mn);
      const float e0 = __expf(s0[j] - mn);
      const float e1 = __expf(s1[j] - mn);
      float rs = e0 + e1;
      rs += __shfl_xor(rs, 1);
      rs += __shfl_xor(rs, 2);
      rs += __shfl_xor(rs, 4);
      rs += __shfl_xor(rs, 8);
      l_r[j] = l_r[j] * sc + rs;
      m_r[j] = mn;
      acc0[j] *= sc;
      acc1[j] *= sc;
      p0[j] = e0;
      p1[j] = e1;
    }

    // C-layout P -> A-layout via per-wave LDS slab (DS ops in-order per wave)
#pragma unroll
    for (int j = 0; j < 8; ++j) {
      const int rloc = j + (hi << 3);
      plds[wave][rloc][l & 15] = (_Float16)p0[j];
      plds[wave][rloc][16 + (l & 15)] = (_Float16)p1[j];
    }
    v16h pf = load_a(&plds[wave][0][0], 32);

    v16h bv0 = load_b(vmat + d0, 256);                      // cc 0..15
    v16h bv1 = load_b(vmat + (size_t)16 * 256 + d0, 256);   // cc 16..31
    acc0 = wmma32(pf, bv0, acc0);
    acc1 = wmma32(pf, bv1, acc1);
  }

#pragma unroll
  for (int j = 0; j < 8; ++j) {
    const int qrow = qbase + j + (hi << 3);
    const float inv = 1.f / l_r[j];
    const int cc0 = l & 15;
    const size_t base = (size_t)(n * 256 + qrow) * 768 + b * 128 + g * 32;
    ACT[base + cc0] = (_Float16)(acc0[j] * inv);
    ACT[base + 16 + cc0] = (_Float16)(acc1[j] * inv);
  }
}

// ---------------------------------------------------------------------------
// LSTM cell pointwise update: gates f32 [m][512] -> h_new f16 [m][128]
// ---------------------------------------------------------------------------
__global__ void k_point(const float* __restrict__ G, const float* __restrict__ btok,
                        const float* __restrict__ c_prev, _Float16* __restrict__ HN,
                        int total) {
  for (int i = blockIdx.x * blockDim.x + threadIdx.x; i < total;
       i += gridDim.x * blockDim.x) {
    const int m = i >> 7, o = i & 127;
    const int nb = m >> 8, hw = m & 255;
    const float* gr = G + (size_t)m * 512;
    const float gi = gr[o] + btok[o];
    const float gf = gr[128 + o] + btok[128 + o];
    const float gg = gr[256 + o] + btok[256 + o];
    const float go = gr[384 + o] + btok[384 + o];
    const float i_ = 1.f / (1.f + __expf(-gi));
    const float f_ = 1.f / (1.f + __expf(-gf));
    const float g_ = tanhf(gg);
    const float o_ = 1.f / (1.f + __expf(-go));
    const float cp = c_prev[((size_t)nb * 128 + o) * 256 + hw];
    const float c = f_ * cp + i_ * g_;
    HN[(size_t)m * 128 + o] = (_Float16)(o_ * tanhf(c));
  }
}

// ---------------------------------------------------------------------------
// Host: launch pipeline
// ---------------------------------------------------------------------------
extern "C" void kernel_launch(void* const* d_in, const int* in_sizes, int n_in,
                              void* d_out, int out_size, void* d_ws, size_t ws_size,
                              hipStream_t stream) {
  (void)in_sizes; (void)n_in; (void)out_size; (void)ws_size;
  const float* x      = (const float*)d_in[0];
  const float* h_prev = (const float*)d_in[1];
  const float* c_prev = (const float*)d_in[2];
  const float* W_in   = (const float*)d_in[3];
  const float* b_in   = (const float*)d_in[4];
  const float* W_cx   = (const float*)d_in[5];
  const float* W_ch   = (const float*)d_in[6];
  const float* W_q    = (const float*)d_in[7];
  const float* W_k    = (const float*)d_in[8];
  const float* W_v    = (const float*)d_in[9];
  const float* W_tok  = (const float*)d_in[10];
  const float* b_tok  = (const float*)d_in[11];
  const float* W_skip = (const float*)d_in[12];
  const float* W_out  = (const float*)d_in[13];
  const float* b_out  = (const float*)d_in[14];

  char* ws = (char*)d_ws;
  auto alloc = [&](size_t bytes) {
    char* p = ws;
    ws += (bytes + 255) & ~(size_t)255;
    return p;
  };
  _Float16* XB    = (_Float16*)alloc((size_t)MTOT * 64 * 2);
  _Float16* HB    = (_Float16*)alloc((size_t)MTOT * 128 * 2);
  _Float16* XT    = (_Float16*)alloc((size_t)MTOT * 128 * 2);
  _Float16* ACT   = (_Float16*)alloc((size_t)MTOT * 768 * 2);   // [a(512)|xc|hc]
  _Float16* QB    = (_Float16*)alloc((size_t)512 * 256 * 32 * 2);
  _Float16* KBuf  = (_Float16*)alloc((size_t)512 * 256 * 32 * 2);
  _Float16* VB    = (_Float16*)alloc((size_t)512 * 32 * 256 * 2);
  float*    Gbuf  = (float*)   alloc((size_t)MTOT * 512 * 4);
  _Float16* HN    = (_Float16*)alloc((size_t)MTOT * 128 * 2);
  _Float16* WBin  = (_Float16*)alloc((size_t)128 * 64 * 2);
  _Float16* WBcx  = (_Float16*)alloc((size_t)128 * 1152 * 2);
  _Float16* WBch  = (_Float16*)alloc((size_t)128 * 1152 * 2);
  _Float16* WBq   = (_Float16*)alloc((size_t)4 * 128 * 128 * 2);
  _Float16* WBk   = (_Float16*)alloc((size_t)4 * 128 * 128 * 2);
  _Float16* WBv   = (_Float16*)alloc((size_t)4 * 128 * 128 * 2);
  _Float16* WG    = (_Float16*)alloc((size_t)512 * 768 * 2);
  _Float16* WBout = (_Float16*)alloc((size_t)128 * 128 * 2);

  // stage activations + weights to f16
  k_cvt_nchw<<<512, 256, 0, stream>>>(x, XB, 64, MTOT * 64);
  k_cvt_nchw<<<1024, 256, 0, stream>>>(h_prev, HB, 128, MTOT * 128);
  k_cvt_flat<<<32, 256, 0, stream>>>(W_in, WBin, 128 * 64);
  k_cvt_conv<<<576, 256, 0, stream>>>(W_cx, WBcx, 128 * 1152);
  k_cvt_conv<<<576, 256, 0, stream>>>(W_ch, WBch, 128 * 1152);
  k_cvt_flat<<<256, 256, 0, stream>>>(W_q, WBq, 4 * 128 * 128);
  k_cvt_flat<<<256, 256, 0, stream>>>(W_k, WBk, 4 * 128 * 128);
  k_cvt_flat<<<256, 256, 0, stream>>>(W_v, WBv, 4 * 128 * 128);
  k_cvt_tok<<<1536, 256, 0, stream>>>(W_tok, W_skip, WG);
  k_cvt_flat<<<64, 256, 0, stream>>>(W_out, WBout, 128 * 128);

  // xt = tanh(x W_in^T + b_in)
  k_gemm<<<dim3(MTOT / 32, 2), 256, 0, stream>>>(XB, 64, WBin, 64, 64, XT, b_in, 0, 0);

  // xc / hc (3x3 convs) -> ACT columns 512.. / 640..
  k_gemm_conv<<<dim3(MTOT / 32, 2), 256, 0, stream>>>(XT, WBcx, ACT, 512);
  k_gemm_conv<<<dim3(MTOT / 32, 2), 256, 0, stream>>>(HB, WBch, ACT, 640);

  // q/k/v projections for 4 branches (q from xx/hh, k,v from x/h/x/h)
  for (int b = 0; b < 4; ++b) {
    const _Float16* qsrc  = (b < 2) ? (ACT + 512) : (ACT + 640);
    const _Float16* kvsrc = (b & 1) ? (ACT + 640) : (ACT + 512);
    k_gemm<<<dim3(MTOT / 32, 2), 256, 0, stream>>>(qsrc, 768, WBq + b * 16384, 128,
                                                   128, QB, nullptr, 2, b);
    k_gemm<<<dim3(MTOT / 32, 2), 256, 0, stream>>>(kvsrc, 768, WBk + b * 16384, 128,
                                                   128, KBuf, nullptr, 2, b);
    k_gemm<<<dim3(MTOT / 32, 2), 256, 0, stream>>>(kvsrc, 768, WBv + b * 16384, 128,
                                                   128, VB, nullptr, 3, b);
  }

  // flash attention -> ACT columns 0..511
  k_attn<<<dim3(2, 512), 256, 0, stream>>>(QB, KBuf, VB, ACT);

  // gates: [8192,768] x [512,768]^T -> f32 [8192,512]
  k_gemm<<<dim3(MTOT / 32, 8), 256, 0, stream>>>(ACT, 768, WG, 768, 768, Gbuf,
                                                 nullptr, 4, 0);

  // LSTM cell update -> h_new (f16)
  k_point<<<2048, 256, 0, stream>>>(Gbuf, b_tok, c_prev, HN, MTOT * 128);

  // output projection -> d_out (f32, NCHW)
  k_gemm<<<dim3(MTOT / 32, 2), 256, 0, stream>>>(HN, 128, WBout, 128, 128, d_out,
                                                 b_out, 5, 0);
}
